// LoRA_qkv_85117661872673
// MI455X (gfx1250) — matrix operand order repaired and checked by reading, not compiled, not run on hardware
//
#include <hip/hip_runtime.h>

// Problem sizes (from reference)
#define DIM   768
#define NOUT  2304      // 3*DIM
#define RANK  16
#define MTOK  32768     // 8*64*64 tokens

// GEMM tiling: 4 waves/block, block tile 128x128, wave tile 64x64, K-step 32
#define BM 128
#define BN 128
#define BK 32
#define KSTEPS (DIM / BK)          // 24
#define LDSS 40                    // LDS row stride in bf16: 80B = 16B-aligned, conflict-free
#define TILE_BYTES (BM * LDSS * 2) // 10240 B per buffer

typedef unsigned short u16;
typedef unsigned int   u32;
typedef unsigned long long u64;
typedef __attribute__((ext_vector_type(16))) __bf16 v16bf;
typedef __attribute__((ext_vector_type(8)))  float  v8f;

static __device__ __forceinline__ u16 f2bf(float f) {
  return __builtin_bit_cast(u16, (__bf16)f);   // v_cvt f32->bf16 (RNE)
}
static __device__ __forceinline__ u32 pack2(float a, float b) {
  return (u32)f2bf(a) | ((u32)f2bf(b) << 16);
}

// Async DMA: global (16B) -> LDS, tracked by ASYNCcnt (CDNA5 ISA)
static __device__ __forceinline__ void async_g2l_b128(u32 lds_off, const void* gptr) {
  asm volatile("global_load_async_to_lds_b128 %0, %1, off"
               :: "v"(lds_off), "v"((u64)(uintptr_t)gptr)
               : "memory");
}
static __device__ __forceinline__ void wait_asynccnt0() {
  asm volatile("s_wait_asynccnt 0" ::: "memory");
}

// ---------- Pass 1: x fp32 -> bf16 ----------
__global__ void cvt_x_bf16(const float* __restrict__ x, uint2* __restrict__ xb, int n4) {
  int i = blockIdx.x * blockDim.x + threadIdx.x;
  if (i >= n4) return;
  float4 f = reinterpret_cast<const float4*>(x)[i];
  uint2 o;
  o.x = pack2(f.x, f.y);
  o.y = pack2(f.z, f.w);
  xb[i] = o;
}

// ---------- Pass 2: W_eff = Wqkv + B_s @ A_s  (folded LoRA), stored bf16 ----------
__global__ void build_weff(const float* __restrict__ W,
                           const float* __restrict__ Aq, const float* __restrict__ Bq,
                           const float* __restrict__ Ak, const float* __restrict__ Bk,
                           const float* __restrict__ Av, const float* __restrict__ Bv,
                           u16* __restrict__ wb) {
  int idx = blockIdx.x * blockDim.x + threadIdx.x;
  if (idx >= NOUT * DIM) return;
  int o = idx / DIM;
  int d = idx - o * DIM;
  int slice = o / DIM;           // 0=q, 1=k, 2=v
  int oo = o - slice * DIM;
  const float* A = (slice == 0) ? Aq : (slice == 1) ? Ak : Av;  // [RANK, DIM]
  const float* B = (slice == 0) ? Bq : (slice == 1) ? Bk : Bv;  // [DIM, RANK]
  float s = W[idx];
#pragma unroll
  for (int r = 0; r < RANK; ++r)
    s += B[oo * RANK + r] * A[r * DIM + d];
  wb[idx] = f2bf(s);
}

// ---------- Pass 3: out[M,NOUT] = Xb @ Wbᵀ + bias, bf16 WMMA, f32 accumulate,
//            double-buffered async-LDS pipeline ----------
__global__ void __launch_bounds__(128)
gemm_qkv(const u16* __restrict__ Xb,   // [MTOK, DIM] bf16
         const u16* __restrict__ Wb,   // [NOUT, DIM] bf16
         const float* __restrict__ bias,
         float* __restrict__ out) {
  __shared__ u16 As[2][BM * LDSS];
  __shared__ u16 Bs[2][BN * LDSS];

  const int tid  = threadIdx.x;
  const int lane = tid & 31;
  const int wave = tid >> 5;        // 0..3
  const int wm   = wave >> 1;       // 0..1 : wave row in block tile
  const int wn   = wave & 1;        // 0..1 : wave col in block tile
  const int half = lane >> 4;       // 0..1
  const int l16  = lane & 15;

  const int n0 = blockIdx.x * BN;
  const int m0 = blockIdx.y * BM;

  // Per-thread staging assignment: 512 16B-chunks per tile, 4 per thread.
  // Precompute global chunk pointers (bump by BK per K-step) and LDS offsets.
  const u16* gA[4];
  const u16* gB[4];
  u32 lofA[2][4], lofB[2][4];
  {
    const u32 asBase = (u32)(uintptr_t)(&As[0][0]);  // generic->LDS: offset in [31:0]
    const u32 bsBase = (u32)(uintptr_t)(&Bs[0][0]);
#pragma unroll
    for (int i = 0; i < 4; ++i) {
      int idx = tid + i * 128;          // 0..511
      int row = idx >> 2;               // 0..127
      int c   = idx & 3;                // 16B chunk within the 64B row slice
      gA[i] = Xb + (size_t)(m0 + row) * DIM + c * 8;
      gB[i] = Wb + (size_t)(n0 + row) * DIM + c * 8;
      u32 l = (u32)(row * LDSS + c * 8) * 2u;  // bytes
      lofA[0][i] = asBase + l;  lofA[1][i] = asBase + TILE_BYTES + l;
      lofB[0][i] = bsBase + l;  lofB[1][i] = bsBase + TILE_BYTES + l;
    }
  }

  // Prologue: DMA tile 0 into buffer 0
#pragma unroll
  for (int i = 0; i < 4; ++i) {
    async_g2l_b128(lofA[0][i], gA[i]);
    async_g2l_b128(lofB[0][i], gB[i]);
    gA[i] += BK;
    gB[i] += BK;
  }
  wait_asynccnt0();
  __syncthreads();

  v8f acc[4][4] = {};

  for (int it = 0; it < KSTEPS; ++it) {
    const int cur = it & 1;

    // ---- issue async DMA for next tile into the other buffer (overlaps compute) ----
    if (it + 1 < KSTEPS) {
      const int nxt = cur ^ 1;
#pragma unroll
      for (int i = 0; i < 4; ++i) {
        async_g2l_b128(lofA[nxt][i], gA[i]);
        async_g2l_b128(lofB[nxt][i], gB[i]);
        __builtin_prefetch(gA[i] + BK, 0, 1);   // speculative: safe past tail
        __builtin_prefetch(gB[i] + BK, 0, 1);
        gA[i] += BK;
        gB[i] += BK;
      }
    }

    // ---- fragment gathers: contiguous k order (k = 16*half + j), same for A and B ----
    v16bf a[4], b[4];
#pragma unroll
    for (int i = 0; i < 4; ++i) {
      // A 16x32 bf16: lane row = l16; elems j: K = 16*half + j (contiguous run of 16)
      const u16* pa = &As[cur][(wm * 64 + i * 16 + l16) * LDSS];
      uint4* av = reinterpret_cast<uint4*>(&a[i]);
      av[0] = *reinterpret_cast<const uint4*>(pa + 16 * half);
      av[1] = *reinterpret_cast<const uint4*>(pa + 16 * half + 8);
      // B 32x16 bf16: lane col = l16; elems j: K = 16*half + j (contiguous run of 16)
      const u16* pb = &Bs[cur][(wn * 64 + i * 16 + l16) * LDSS + 16 * half];
      uint4* bv = reinterpret_cast<uint4*>(&b[i]);
      bv[0] = *reinterpret_cast<const uint4*>(pb);
      bv[1] = *reinterpret_cast<const uint4*>(pb + 8);
    }

    // ---- 16 WMMAs per K-step (64x64 wave tile); DMA for tile it+1 runs underneath ----
#pragma unroll
    for (int i = 0; i < 4; ++i)
#pragma unroll
      for (int j = 0; j < 4; ++j)
        acc[i][j] = __builtin_amdgcn_wmma_f32_16x16x32_bf16(
            false, a[i], false, b[j], (short)0, acc[i][j], false, false);

    // next buffer must be fully written (own DMA done + all waves arrived)
    wait_asynccnt0();
    __syncthreads();
  }

  // ---- epilogue: C/D layout -> global, add bias; NT stores (out is write-once,
  //      keep the 48MB bf16 x panel resident in the 192MB L2) ----
#pragma unroll
  for (int j = 0; j < 4; ++j) {
    int col = n0 + wn * 64 + j * 16 + l16;
    float bj = bias[col];
#pragma unroll
    for (int i = 0; i < 4; ++i) {
      int rbase = m0 + wm * 64 + i * 16 + half * 8;   // M = vgpr + 8*half
#pragma unroll
      for (int e = 0; e < 8; ++e)
        __builtin_nontemporal_store(acc[i][j][e] + bj,
                                    out + (size_t)(rbase + e) * NOUT + col);
    }
  }
}

extern "C" void kernel_launch(void* const* d_in, const int* in_sizes, int n_in,
                              void* d_out, int out_size, void* d_ws, size_t ws_size,
                              hipStream_t stream) {
  const float* x    = (const float*)d_in[0];
  const float* Wqkv = (const float*)d_in[1];
  const float* bqkv = (const float*)d_in[2];
  const float* Aq   = (const float*)d_in[3];
  const float* Bq   = (const float*)d_in[4];
  const float* Ak   = (const float*)d_in[5];
  const float* Bk   = (const float*)d_in[6];
  const float* Av   = (const float*)d_in[7];
  const float* Bv   = (const float*)d_in[8];

  // Workspace layout: x_bf16 [MTOK*DIM] (48 MB) | W_eff_bf16 [NOUT*DIM] (3.5 MB)
  u16* xb = (u16*)d_ws;
  u16* wb = xb + (size_t)MTOK * DIM;

  int n4 = MTOK * DIM / 4;
  cvt_x_bf16<<<(n4 + 255) / 256, 256, 0, stream>>>(x, (uint2*)xb, n4);

  int nw = NOUT * DIM;
  build_weff<<<(nw + 255) / 256, 256, 0, stream>>>(Wqkv, Aq, Bq, Ak, Bk, Av, Bv, wb);

  dim3 grid(NOUT / BN, MTOK / BM);   // 18 x 256 blocks
  gemm_qkv<<<grid, 128, 0, stream>>>(xb, wb, bqkv, (float*)d_out);
}
